// SparseSiLUFFN_38242388803683
// MI455X (gfx1250) — compile-verified, matrix operand order checked by reading, not run on hardware
//
#include <hip/hip_runtime.h>
#include <hip/hip_bf16.h>

typedef __attribute__((ext_vector_type(8)))  float  v8f;
typedef __attribute__((ext_vector_type(8)))  __bf16 v8bf;
typedef __attribute__((ext_vector_type(16))) __bf16 v16bf;

#define D_MODEL 2048
#define D_FFN   8192
#define TOP_K   256

// ---------- helpers ----------
static __device__ __forceinline__ unsigned short f2bf(float f) {
    unsigned u = __float_as_uint(f);
    u += 0x7FFFu + ((u >> 16) & 1u);   // round-to-nearest-even
    return (unsigned short)(u >> 16);
}
static __device__ __forceinline__ float bf2f(unsigned short h) {
    return __uint_as_float(((unsigned)h) << 16);
}
// monotone 16-bit key: larger float -> larger key
static __device__ __forceinline__ int key16(unsigned short h) {
    return (h & 0x8000u) ? (int)((~h) & 0xFFFFu) : (int)(h | 0x8000u);
}

// ---------- 1) f32 -> bf16 convert (x) ----------
__global__ __launch_bounds__(256) void convert_f32_bf16(const float* __restrict__ src,
                                                        unsigned short* __restrict__ dst,
                                                        long n) {
    long i = ((long)blockIdx.x * blockDim.x + threadIdx.x) * 4;
    if (i >= n) return;
    float4 f = *(const float4*)(src + i);
    uint2 o;
    o.x = (unsigned)f2bf(f.x) | ((unsigned)f2bf(f.y) << 16);
    o.y = (unsigned)f2bf(f.z) | ((unsigned)f2bf(f.w) << 16);
    *(uint2*)(dst + i) = o;
}

// ---------- 2) f32 [R][C] -> bf16 transpose [C][R] ----------
__global__ __launch_bounds__(256) void transpose_f32_bf16(const float* __restrict__ src,
                                                          unsigned short* __restrict__ dst,
                                                          int R, int C) {
    __shared__ float tile[32][33];
    int cb = blockIdx.x * 32, rb = blockIdx.y * 32;
    int tx = threadIdx.x, ty = threadIdx.y;
#pragma unroll
    for (int i = 0; i < 4; ++i)
        tile[ty + i * 8][tx] = src[(size_t)(rb + ty + i * 8) * C + cb + tx];
    __syncthreads();
#pragma unroll
    for (int i = 0; i < 4; ++i) {
        int cl = ty + i * 8;
        dst[(size_t)(cb + cl) * R + rb + tx] = f2bf(tile[tx][cl]);
    }
}

// ---------- 3) bf16 WMMA GEMM:  C[M][N] = A[M][K] * BT[N][K]^T ----------
// 256x128 block tile, 8 waves x (32 rows x 128 cols): 16 WMMAs per k-step per
// wave, each B fragment reused for two row-tiles (half the LDS traffic/FLOP).
// B tile staged LDS-direct via global_load_async_to_lds_b128 (ASYNCcnt),
// double-buffered: one s_wait_asynccnt + one barrier per k-step; batched
// ds_load_b128 groups with a single s_wait_dscnt inside asm so WMMAs issue
// back-to-back instead of waiting on LDS per-WMMA.
__global__ __launch_bounds__(256) void gemm_bf16_wmma(const unsigned short* __restrict__ A,
                                                      const unsigned short* __restrict__ BT,
                                                      void* __restrict__ C,
                                                      int M, int N, int K, int c_bf16) {
    __shared__ __align__(16) unsigned short Bs[2][128 * 40]; // stride 40 -> conflict-free b128 reads

    const int lane = threadIdx.x & 31;
    const int wave = threadIdx.x >> 5;
    const int n0   = blockIdx.x * 128;
    const int m0   = blockIdx.y * 256 + wave * 32;
    const int row  = lane & 15;
    const int koff = (lane >> 4) * 8;   // ISA 16-bit fragment: K halves 0..7 / 8..15

    const unsigned short* arow0 = A + (size_t)(m0 + row) * K;
    const unsigned short* arow1 = A + (size_t)(m0 + 16 + row) * K;

    // async B staging: thread -> (row n0+nl, 32B segment q)
    const int nl = threadIdx.x >> 1;          // 0..127
    const int q  = (threadIdx.x & 1) * 16;    // element offset 0 or 16
    const unsigned short* bsrc = BT + (size_t)(n0 + nl) * K + q;
    const unsigned lds0 = (unsigned)(uintptr_t)&Bs[0][nl * 40 + q];
    const unsigned lds1 = (unsigned)(uintptr_t)&Bs[1][nl * 40 + q];

    // prologue: stage tile 0 into buffer 0 (two b128 = 32B per lane)
    {
        unsigned long long ga = (unsigned long long)(uintptr_t)bsrc;
        asm volatile("global_load_async_to_lds_b128 %0, %1, off\n\t"
                     "global_load_async_to_lds_b128 %0, %1, off offset:16"
                     :: "v"(lds0), "v"(ga) : "memory");
    }

    v8f acc[2][8] = {};
    int buf = 0;
    for (int k = 0; k < K; k += 32, buf ^= 1) {
        asm volatile("s_wait_asynccnt 0x0" ::: "memory"); // my tile-k writes landed
        __syncthreads();                                  // everyone's landed; prev reads done

        if (k + 32 < K) {  // overlap: stage tile k+1 into the other buffer
            unsigned long long ga = (unsigned long long)(uintptr_t)(bsrc + k + 32);
            unsigned ldsn = buf ? lds0 : lds1;
            asm volatile("global_load_async_to_lds_b128 %0, %1, off\n\t"
                         "global_load_async_to_lds_b128 %0, %1, off offset:16"
                         :: "v"(ldsn), "v"(ga) : "memory");
        }

        // two A fragments (16x32 bf16 each): contiguous 16B runs per lane
        v8bf a0l = *(const v8bf*)(arow0 + k + koff);
        v8bf a0h = *(const v8bf*)(arow0 + k + koff + 16);
        v8bf a1l = *(const v8bf*)(arow1 + k + koff);
        v8bf a1h = *(const v8bf*)(arow1 + k + koff + 16);
        v16bf af0 = __builtin_shufflevector(a0l, a0h, 0, 1, 2, 3, 4, 5, 6, 7,
                                            8, 9, 10, 11, 12, 13, 14, 15);
        v16bf af1 = __builtin_shufflevector(a1l, a1h, 0, 1, 2, 3, 4, 5, 6, 7,
                                            8, 9, 10, 11, 12, 13, 14, 15);
        if (k + 64 < K) {
            __builtin_prefetch(arow0 + k + 64, 0, 3);
            __builtin_prefetch(arow1 + k + 64, 0, 3);
        }

        const unsigned baddr = (unsigned)(uintptr_t)&Bs[buf][row * 40 + koff];
        v8bf bl[8];

        // ---- group 0: B fragments j=0..3, batched loads + single dscnt wait ----
        asm volatile(
            "ds_load_b128 %0, %8\n\t"
            "ds_load_b128 %1, %8 offset:32\n\t"
            "ds_load_b128 %2, %8 offset:1280\n\t"
            "ds_load_b128 %3, %8 offset:1312\n\t"
            "ds_load_b128 %4, %8 offset:2560\n\t"
            "ds_load_b128 %5, %8 offset:2592\n\t"
            "ds_load_b128 %6, %8 offset:3840\n\t"
            "ds_load_b128 %7, %8 offset:3872\n\t"
            "s_wait_dscnt 0x0"
            : "=v"(bl[0]), "=v"(bl[1]), "=v"(bl[2]), "=v"(bl[3]),
              "=v"(bl[4]), "=v"(bl[5]), "=v"(bl[6]), "=v"(bl[7])
            : "v"(baddr)
            : "memory");
#pragma unroll
        for (int j = 0; j < 4; ++j) {
            v16bf bfr = __builtin_shufflevector(bl[2 * j], bl[2 * j + 1],
                                                0, 1, 2, 3, 4, 5, 6, 7,
                                                8, 9, 10, 11, 12, 13, 14, 15);
            acc[0][j] = __builtin_amdgcn_wmma_f32_16x16x32_bf16(
                false, af0, false, bfr, (short)0, acc[0][j], false, false);
            acc[1][j] = __builtin_amdgcn_wmma_f32_16x16x32_bf16(
                false, af1, false, bfr, (short)0, acc[1][j], false, false);
        }

        // ---- group 1: B fragments j=4..7 ----
        asm volatile(
            "ds_load_b128 %0, %8 offset:5120\n\t"
            "ds_load_b128 %1, %8 offset:5152\n\t"
            "ds_load_b128 %2, %8 offset:6400\n\t"
            "ds_load_b128 %3, %8 offset:6432\n\t"
            "ds_load_b128 %4, %8 offset:7680\n\t"
            "ds_load_b128 %5, %8 offset:7712\n\t"
            "ds_load_b128 %6, %8 offset:8960\n\t"
            "ds_load_b128 %7, %8 offset:8992\n\t"
            "s_wait_dscnt 0x0"
            : "=v"(bl[0]), "=v"(bl[1]), "=v"(bl[2]), "=v"(bl[3]),
              "=v"(bl[4]), "=v"(bl[5]), "=v"(bl[6]), "=v"(bl[7])
            : "v"(baddr)
            : "memory");
#pragma unroll
        for (int j = 0; j < 4; ++j) {
            v16bf bfr = __builtin_shufflevector(bl[2 * j], bl[2 * j + 1],
                                                0, 1, 2, 3, 4, 5, 6, 7,
                                                8, 9, 10, 11, 12, 13, 14, 15);
            acc[0][4 + j] = __builtin_amdgcn_wmma_f32_16x16x32_bf16(
                false, af0, false, bfr, (short)0, acc[0][4 + j], false, false);
            acc[1][4 + j] = __builtin_amdgcn_wmma_f32_16x16x32_bf16(
                false, af1, false, bfr, (short)0, acc[1][4 + j], false, false);
        }
        // no trailing barrier: next iteration's wait+barrier fences buffer reuse
    }

    // epilogue: C/D layout -> lanes 0-15: M=v, lanes 16-31: M=v+8
    const int rh = (lane >> 4) * 8;
    const int cn = lane & 15;
    if (c_bf16) {
        unsigned short* Cb = (unsigned short*)C;
#pragma unroll
        for (int r = 0; r < 2; ++r)
#pragma unroll
            for (int j = 0; j < 8; ++j)
#pragma unroll
                for (int v = 0; v < 8; ++v)
                    Cb[(size_t)(m0 + r * 16 + rh + v) * N + n0 + j * 16 + cn] = f2bf(acc[r][j][v]);
    } else {
        float* Cf = (float*)C;
#pragma unroll
        for (int r = 0; r < 2; ++r)
#pragma unroll
            for (int j = 0; j < 8; ++j)
#pragma unroll
                for (int v = 0; v < 8; ++v)
                    Cf[(size_t)(m0 + r * 16 + rh + v) * N + n0 + j * 16 + cn] = acc[r][j][v];
    }
}

// ---------- 4) deterministic top-256 + fused SiLU*U -> compact Z, idx ----------
__global__ __launch_bounds__(256) void topk_silu_kernel(const unsigned short* __restrict__ G,
                                                        const unsigned short* __restrict__ U,
                                                        float* __restrict__ Zc,
                                                        int* __restrict__ idxo) {
    __shared__ unsigned short gs[D_FFN];   // 16 KB row cache
    __shared__ int hist[256];
    __shared__ int scn[512];
    __shared__ int ctrl[8];

    const int t = threadIdx.x;
    const int n = blockIdx.x;
    const unsigned short* grow = G + (size_t)n * D_FFN;

    hist[t] = 0;
    __syncthreads();

    // pass 1: cache row in LDS + hi-byte histogram of keys
#pragma unroll
    for (int it = 0; it < 4; ++it) {
        int base = it * 2048 + t * 8;
        uint4 d = *(const uint4*)(grow + base);
        *(uint4*)&gs[base] = d;
        unsigned w[4] = {d.x, d.y, d.z, d.w};
#pragma unroll
        for (int c = 0; c < 4; ++c) {
            atomicAdd(&hist[key16((unsigned short)(w[c] & 0xFFFFu)) >> 8], 1);
            atomicAdd(&hist[key16((unsigned short)(w[c] >> 16)) >> 8], 1);
        }
    }
    __syncthreads();

    if (t == 0) {  // threshold hi-byte bin (descending)
        int cum = 0, b = 255;
        for (; b > 0; --b) {
            if (cum + hist[b] >= TOP_K) break;
            cum += hist[b];
        }
        ctrl[0] = b;
        ctrl[1] = cum;  // strictly above bin b
    }
    __syncthreads();
    const int bsel = ctrl[0];
    const int cumAbove = ctrl[1];

    hist[t] = 0;
    __syncthreads();
    // pass 2: lo-byte histogram within threshold bin
    for (int i = t; i < D_FFN; i += 256) {
        int key = key16(gs[i]);
        if ((key >> 8) == bsel) atomicAdd(&hist[key & 255], 1);
    }
    __syncthreads();

    if (t == 0) {
        int need = TOP_K - cumAbove;
        int cum = 0, l = 255;
        for (; l > 0; --l) {
            if (cum + hist[l] >= need) break;
            cum += hist[l];
        }
        int nAbove = cumAbove + cum;
        ctrl[2] = (bsel << 8) | l;   // exact 16-bit threshold key T
        ctrl[3] = nAbove;
        ctrl[4] = TOP_K - nAbove;    // #ties taken, lowest index first
    }
    __syncthreads();
    const int T = ctrl[2], nAbove = ctrl[3], nEqNeed = ctrl[4];

    // pass 3: index-ordered chunk scan (thread t owns [t*32, t*32+32))
    int ca = 0, ce = 0;
#pragma unroll
    for (int i = 0; i < 32; ++i) {
        int key = key16(gs[t * 32 + i]);
        if (key > T) ca++;
        else if (key == T) ce++;
    }
    scn[t] = ca;
    scn[256 + t] = ce;
    __syncthreads();
    if (t == 0) {  // serial exclusive scans -> deterministic slots
        int sa = 0, se = 0;
        for (int i = 0; i < 256; ++i) {
            int a = scn[i];       scn[i] = sa;       sa += a;
            int e = scn[256 + i]; scn[256 + i] = se; se += e;
        }
    }
    __syncthreads();
    int pa = scn[t], pe = scn[256 + t];
#pragma unroll 1
    for (int i = 0; i < 32; ++i) {
        int e = t * 32 + i;
        unsigned short h = gs[e];
        int key = key16(h);
        int slot = -1;
        if (key > T) slot = pa++;
        else if (key == T) { if (pe < nEqNeed) slot = nAbove + pe; pe++; }
        if (slot >= 0) {
            float g = bf2f(h);
            float u = bf2f(U[(size_t)n * D_FFN + e]);
            float s = g / (1.0f + __expf(-g));  // SiLU
            Zc[(size_t)n * TOP_K + slot] = s * u;
            idxo[(size_t)n * TOP_K + slot] = e;
        }
    }
}

// ---------- 5) zero + scatter compact Z into dense bf16 Zd ----------
__global__ __launch_bounds__(256) void scatter_kernel(const float* __restrict__ Zc,
                                                      const int* __restrict__ idx,
                                                      unsigned short* __restrict__ Zd) {
    const int n = blockIdx.x, t = threadIdx.x;
    unsigned short* row = Zd + (size_t)n * D_FFN;
    uint4 z; z.x = z.y = z.z = z.w = 0u;
#pragma unroll
    for (int i = 0; i < 4; ++i) *(uint4*)&row[(t + i * 256) * 8] = z;
    __syncthreads();
    float zv = Zc[(size_t)n * TOP_K + t];
    int e = idx[(size_t)n * TOP_K + t];
    row[e] = f2bf(zv);
}

// ---------- launch ----------
extern "C" void kernel_launch(void* const* d_in, const int* in_sizes, int n_in,
                              void* d_out, int out_size, void* d_ws, size_t ws_size,
                              hipStream_t stream) {
    const float* x  = (const float*)d_in[0];
    const float* wg = (const float*)d_in[1];
    const float* wu = (const float*)d_in[2];
    const float* wd = (const float*)d_in[3];
    const int Ntok = in_sizes[0] / D_MODEL;  // 8192

    char* ws = (char*)d_ws;
    size_t off = 0;
    unsigned short* xb  = (unsigned short*)(ws + off); off += (size_t)Ntok * D_MODEL * 2;
    unsigned short* wgT = (unsigned short*)(ws + off); off += (size_t)D_MODEL * D_FFN * 2;
    unsigned short* wuT = (unsigned short*)(ws + off); off += (size_t)D_MODEL * D_FFN * 2;
    unsigned short* wdT = (unsigned short*)(ws + off); off += (size_t)D_FFN * D_MODEL * 2;
    unsigned short* G   = (unsigned short*)(ws + off); off += (size_t)Ntok * D_FFN * 2;
    unsigned short* U   = (unsigned short*)(ws + off); off += (size_t)Ntok * D_FFN * 2;
    unsigned short* Zd  = G;  // G dead after top-k; reuse
    float* Zc = (float*)(ws + off); off += (size_t)Ntok * TOP_K * 4;
    int*   idx = (int*)(ws + off);  off += (size_t)Ntok * TOP_K * 4;

    long nx = (long)Ntok * D_MODEL;
    convert_f32_bf16<<<(unsigned)((nx / 4 + 255) / 256), 256, 0, stream>>>(x, xb, nx);
    transpose_f32_bf16<<<dim3(D_FFN / 32, D_MODEL / 32), dim3(32, 8), 0, stream>>>(wg, wgT, D_MODEL, D_FFN);
    transpose_f32_bf16<<<dim3(D_FFN / 32, D_MODEL / 32), dim3(32, 8), 0, stream>>>(wu, wuT, D_MODEL, D_FFN);
    transpose_f32_bf16<<<dim3(D_MODEL / 32, D_FFN / 32), dim3(32, 8), 0, stream>>>(wd, wdT, D_FFN, D_MODEL);

    gemm_bf16_wmma<<<dim3(D_FFN / 128, Ntok / 256), 256, 0, stream>>>(xb, wgT, G, Ntok, D_FFN, D_MODEL, 1);
    gemm_bf16_wmma<<<dim3(D_FFN / 128, Ntok / 256), 256, 0, stream>>>(xb, wuT, U, Ntok, D_FFN, D_MODEL, 1);

    topk_silu_kernel<<<Ntok, 256, 0, stream>>>(G, U, Zc, idx);
    scatter_kernel<<<Ntok, 256, 0, stream>>>(Zc, idx, Zd);

    gemm_bf16_wmma<<<dim3(D_MODEL / 128, Ntok / 256), 256, 0, stream>>>(Zd, wdT, (float*)d_out, Ntok, D_MODEL, D_FFN, 0);
}
